// PointNetPP_54700703482023
// MI455X (gfx1250) — matrix operand order repaired and checked by reading, not compiled, not run on hardware
//
#include <hip/hip_runtime.h>
#include <hip/hip_bf16.h>

// Problem constants (match reference)
#define BB     8
#define NN     4096
#define MM     1024
#define KK     16
#define R2C    0.09f      // R*R, R=0.3
#define CIN    3
#define HID    64
#define COUT   128
#define SLOTS  17         // K neighbors + 1 self loop

typedef _Float16 v16h __attribute__((ext_vector_type(16)));
typedef _Float16 v8h  __attribute__((ext_vector_type(8)));
typedef float    v8f  __attribute__((ext_vector_type(8)));

// Order-preserving float <-> uint mapping so masked float max == uint atomic max.
__device__ __forceinline__ unsigned encOrderedF32(float f) {
  unsigned u = __float_as_uint(f);
  return (u & 0x80000000u) ? ~u : (u | 0x80000000u);
}
__device__ __forceinline__ float decOrderedF32(unsigned e) {
  return (e & 0x80000000u) ? __uint_as_float(e & 0x7fffffffu)
                           : __uint_as_float(~e);
}
// enc(-inf) = ~0xFF800000 = 0x007FFFFF
#define ENC_NEG_INF 0x007FFFFFu

// Load a 16x32 f16 A-fragment row slice: lane L (kh = 8*(L>=16))
// needs K in [2*kh,2*kh+16) (matches B's contiguous order) -> two 16B LDS loads.
__device__ __forceinline__ v16h loadAfrag(const _Float16* rowPtr, int kh) {
  const v8h lo = *(const v8h*)(rowPtr + 2 * kh);
  const v8h hi = *(const v8h*)(rowPtr + 2 * kh + 8);
  v16h a;
#pragma unroll
  for (int i = 0; i < 8; ++i) { a[i] = lo[i]; a[8 + i] = hi[i]; }
  return a;
}

// ---------------------------------------------------------------------------
// 1) Farthest point sampling: one block (1024 thr = 32 wave32) per cloud.
//    Distances live in registers (4 points/thread); argmax via shuffle + LDS.
// ---------------------------------------------------------------------------
__global__ __launch_bounds__(1024) void fps_kernel(const float* __restrict__ pos,
                                                   int* __restrict__ fpsIdx) {
  const int b = blockIdx.x;
  const int t = threadIdx.x;
  __shared__ float sVal[32];
  __shared__ int   sIdx[32];
  __shared__ int   sWin;

  const float* pb = pos + (size_t)b * NN * 3;
  float px[4], py[4], pz[4], d[4];
  const float x0 = pb[0], y0 = pb[1], z0 = pb[2];
#pragma unroll
  for (int j = 0; j < 4; ++j) {
    const int p = t + j * 1024;
    px[j] = pb[p * 3 + 0];
    py[j] = pb[p * 3 + 1];
    pz[j] = pb[p * 3 + 2];
    const float dx = px[j] - x0, dy = py[j] - y0, dz = pz[j] - z0;
    d[j] = dx * dx + dy * dy + dz * dz;
  }
  if (t == 0) fpsIdx[b * MM] = 0;

  for (int s = 1; s < MM; ++s) {
    float bv = d[0]; int bi = t;
#pragma unroll
    for (int j = 1; j < 4; ++j)
      if (d[j] > bv) { bv = d[j]; bi = t + j * 1024; }
#pragma unroll
    for (int off = 16; off > 0; off >>= 1) {
      const float ov = __shfl_down(bv, off);
      const int   oi = __shfl_down(bi, off);
      if (ov > bv) { bv = ov; bi = oi; }
    }
    if ((t & 31) == 0) { sVal[t >> 5] = bv; sIdx[t >> 5] = bi; }
    __syncthreads();
    if (t < 32) {
      bv = sVal[t]; bi = sIdx[t];
#pragma unroll
      for (int off = 16; off > 0; off >>= 1) {
        const float ov = __shfl_down(bv, off);
        const int   oi = __shfl_down(bi, off);
        if (ov > bv) { bv = ov; bi = oi; }
      }
      if (t == 0) { sWin = bi; fpsIdx[b * MM + s] = bi; }
    }
    __syncthreads();
    const int w = sWin;
    const float wx = pb[w * 3 + 0], wy = pb[w * 3 + 1], wz = pb[w * 3 + 2];
#pragma unroll
    for (int j = 0; j < 4; ++j) {
      const float dx = px[j] - wx, dy = py[j] - wy, dz = pz[j] - wz;
      d[j] = fminf(d[j], dx * dx + dy * dy + dz * dz);
    }
  }
}

// ---------------------------------------------------------------------------
// 2) Radius search: one wave32 per query; ballot + first-K-by-index selection.
// ---------------------------------------------------------------------------
__global__ __launch_bounds__(256) void radius_kernel(const float* __restrict__ pos,
                                                     const int* __restrict__ fpsIdx,
                                                     int* __restrict__ neigh,
                                                     unsigned* __restrict__ maskbits,
                                                     float* __restrict__ pos_s) {
  const int wid = (int)((blockIdx.x * blockDim.x + threadIdx.x) >> 5);
  const int L = threadIdx.x & 31;
  if (wid >= BB * MM) return;
  const int q = wid;
  const int b = q >> 10;        // q / MM
  const int m = q & 1023;       // q % MM
  const int sIdx = fpsIdx[q];
  const float* ps = pos + ((size_t)b * NN + sIdx) * 3;
  const float qx = ps[0], qy = ps[1], qz = ps[2];
  if (L == 0) { pos_s[q * 3 + 0] = qx; pos_s[q * 3 + 1] = qy; pos_s[q * 3 + 2] = qz; }

  int cnt = 0;
  unsigned mk = 0;
  const long long dstFlat = (long long)b * MM + m;   // PyG flat self index
  for (int base = 0; base < NN; base += 32) {
    if (cnt >= KK) break;
    const int p = base + L;
    const float* pp = pos + ((size_t)b * NN + p) * 3;
    const float dx = pp[0] - qx, dy = pp[1] - qy, dz = pp[2] - qz;
    bool valid = (dx * dx + dy * dy + dz * dz) <= R2C;
    const long long srcFlat = (long long)b * NN + p;
    valid = valid && (srcFlat != dstFlat);           // remove numeric self loop
    unsigned bm = (unsigned)__ballot(valid);
    while (bm && cnt < KK) {
      const int bit = __ffs(bm) - 1;
      bm &= bm - 1;
      if (L == 0) neigh[q * KK + cnt] = base + bit;
      mk |= 1u << cnt;
      ++cnt;
    }
  }
  if (L == 0) {
    for (int c = cnt; c < KK; ++c) neigh[q * KK + c] = 0;
    maskbits[q] = mk;
  }
}

// ---------------------------------------------------------------------------
// 3) Init output accumulator to enc(-inf)
// ---------------------------------------------------------------------------
__global__ void init_kernel(unsigned* __restrict__ outMax) {
  const int i = blockIdx.x * 256 + threadIdx.x;
  if (i < BB * MM * COUT) outMax[i] = ENC_NEG_INF;
}

// ---------------------------------------------------------------------------
// 4) Fused MLP: 64 edge-rows per workgroup (4 waves x 16-row tiles), f16
//    WMMA 16x16x32 with f32 accumulate, masked uint-atomic-max aggregation.
//    Weights staged TRANSPOSED [N][K] so a B-fragment is one 32B LDS read;
//    activations row-major [row][K] so an A-fragment is two 16B LDS reads.
// ---------------------------------------------------------------------------
__global__ __launch_bounds__(128) void mlp_kernel(
    const float* __restrict__ x, const float* __restrict__ pos,
    const float* __restrict__ W1, const float* __restrict__ b1,
    const float* __restrict__ W2, const float* __restrict__ b2,
    const float* __restrict__ W3, const float* __restrict__ b3,
    const int* __restrict__ neigh, const unsigned* __restrict__ maskbits,
    const float* __restrict__ pos_s, unsigned* __restrict__ outMax) {
  __shared__ alignas(32) _Float16 sFeat[64][32];    // K padded 6 -> 32
  __shared__ alignas(32) _Float16 sH1[64][HID];
  __shared__ alignas(32) _Float16 sH2[64][HID];
  __shared__ alignas(32) _Float16 sW1t[HID][32];    // [n][k], k rows 6..31 zero
  __shared__ alignas(32) _Float16 sW2t[HID][HID];   // [n][k]
  __shared__ alignas(32) _Float16 sW3t[COUT][HID];  // [n][k]
  __shared__ float sB1[HID], sB2[HID], sB3[COUT];
  __shared__ int sQ[64];
  __shared__ unsigned char sValid[64];

  const int t = threadIdx.x;
  if (t == 0) { __builtin_prefetch(W2, 0, 0); __builtin_prefetch(W3, 0, 0); }

  // Stage weights transposed (f32 -> f16) into LDS
  for (int i = t; i < HID * 32; i += 128) {
    const int n = i >> 5, k = i & 31;
    sW1t[n][k] = (_Float16)((k < CIN + 3) ? W1[k * HID + n] : 0.0f);
  }
  for (int i = t; i < HID * HID; i += 128) {
    const int n = i >> 6, k = i & 63;
    sW2t[n][k] = (_Float16)W2[k * HID + n];
  }
  for (int i = t; i < COUT * HID; i += 128) {
    const int n = i >> 6, k = i & 63;
    sW3t[n][k] = (_Float16)W3[k * COUT + n];
  }
  if (t < HID) { sB1[t] = b1[t]; sB2[t] = b2[t]; }
  sB3[t] = b3[t];   // 128 threads == COUT

  // Build 64 feature rows [x_j(3), rel(3)] padded with zeros
  const int rowBase = blockIdx.x * 64;
  if (t < 64) {
    const int r = rowBase + t;
    const int q = r / SLOTS;
    const int s = r - q * SLOTS;
    const int b = q >> 10;
    const float qx = pos_s[q * 3 + 0], qy = pos_s[q * 3 + 1], qz = pos_s[q * 3 + 2];
    float fx = 0.f, fy = 0.f, fz = 0.f, rx = 0.f, ry = 0.f, rz = 0.f;
    int valid = 0;
    if (s == SLOTS - 1) {
      // PyG self loop: flat numeric index i_self == q (indexes into x/pos of B*N rows)
      fx = x[q * 3 + 0]; fy = x[q * 3 + 1]; fz = x[q * 3 + 2];
      rx = pos[q * 3 + 0] - qx; ry = pos[q * 3 + 1] - qy; rz = pos[q * 3 + 2] - qz;
      valid = 1;
    } else if ((maskbits[q] >> s) & 1u) {
      const int p = neigh[q * KK + s];
      const size_t g = (size_t)b * NN + p;
      fx = x[g * 3 + 0]; fy = x[g * 3 + 1]; fz = x[g * 3 + 2];
      rx = pos[g * 3 + 0] - qx; ry = pos[g * 3 + 1] - qy; rz = pos[g * 3 + 2] - qz;
      valid = 1;
    }
    sFeat[t][0] = (_Float16)fx; sFeat[t][1] = (_Float16)fy; sFeat[t][2] = (_Float16)fz;
    sFeat[t][3] = (_Float16)rx; sFeat[t][4] = (_Float16)ry; sFeat[t][5] = (_Float16)rz;
#pragma unroll
    for (int c = CIN + 3; c < 32; ++c) sFeat[t][c] = (_Float16)0.0f;
    sQ[t] = q;
    sValid[t] = (unsigned char)valid;
  }
  __syncthreads();

  const int wave   = t >> 5;
  const int L      = t & 31;
  const int rB     = wave * 16;       // this wave's 16-row tile
  const int rowSel = L & 15;          // A-matrix: M = lane % 16
  const int kh     = (L >> 4) * 8;    // A-matrix: +16 K (2*kh) for upper lane half
  const int rOff   = (L >> 4) * 8;    // C/D: +8 M for upper lane half
  const int col16  = L & 15;          // B & C/D: N = lane % 16
  const int kbHalf = (L >> 4) * 16;   // B-matrix: +16 K for upper lane half

  // ---- Layer 1: feats(16x32) @ W1(32x64) ----
  const v16h a1 = loadAfrag(&sFeat[rB + rowSel][0], kh);
#pragma unroll
  for (int n = 0; n < 4; ++n) {
    const v16h bf = *(const v16h*)&sW1t[n * 16 + col16][kbHalf];
    v8f c = {};
    c = __builtin_amdgcn_wmma_f32_16x16x32_f16(false, a1, false, bf, (short)0, c,
                                               false, false);
    const float bias = sB1[n * 16 + col16];
#pragma unroll
    for (int v = 0; v < 8; ++v) {
      float h = c[v] + bias;
      h = h > 0.f ? h : 0.f;
      sH1[rB + v + rOff][n * 16 + col16] = (_Float16)h;
    }
  }
  __syncthreads();

  // ---- Layer 2: h1(16x64) @ W2(64x64), K split in two 32-chunks ----
  const v16h a2lo = loadAfrag(&sH1[rB + rowSel][0], kh);        // K [0,32)
  const v16h a2hi = loadAfrag(&sH1[rB + rowSel][0] + 32, kh);   // K [32,64)
#pragma unroll
  for (int n = 0; n < 4; ++n) {
    const v16h bf0 = *(const v16h*)&sW2t[n * 16 + col16][kbHalf];
    const v16h bf1 = *(const v16h*)&sW2t[n * 16 + col16][32 + kbHalf];
    v8f c = {};
    c = __builtin_amdgcn_wmma_f32_16x16x32_f16(false, a2lo, false, bf0, (short)0, c,
                                               false, false);
    c = __builtin_amdgcn_wmma_f32_16x16x32_f16(false, a2hi, false, bf1, (short)0, c,
                                               false, false);
    const float bias = sB2[n * 16 + col16];
#pragma unroll
    for (int v = 0; v < 8; ++v) {
      float h = c[v] + bias;
      h = h > 0.f ? h : 0.f;
      sH2[rB + v + rOff][n * 16 + col16] = (_Float16)h;
    }
  }
  __syncthreads();

  // ---- Layer 3: h2(16x64) @ W3(64x128) + masked atomic max aggregation ----
  const v16h a3lo = loadAfrag(&sH2[rB + rowSel][0], kh);        // K [0,32)
  const v16h a3hi = loadAfrag(&sH2[rB + rowSel][0] + 32, kh);   // K [32,64)
#pragma unroll
  for (int n = 0; n < 8; ++n) {
    const v16h bf0 = *(const v16h*)&sW3t[n * 16 + col16][kbHalf];
    const v16h bf1 = *(const v16h*)&sW3t[n * 16 + col16][32 + kbHalf];
    v8f c = {};
    c = __builtin_amdgcn_wmma_f32_16x16x32_f16(false, a3lo, false, bf0, (short)0, c,
                                               false, false);
    c = __builtin_amdgcn_wmma_f32_16x16x32_f16(false, a3hi, false, bf1, (short)0, c,
                                               false, false);
    const float bias = sB3[n * 16 + col16];
#pragma unroll
    for (int v = 0; v < 8; ++v) {
      const int lr = rB + v + rOff;
      if (sValid[lr]) {
        const float h = c[v] + bias;
        atomicMax(&outMax[(size_t)sQ[lr] * COUT + n * 16 + col16], encOrderedF32(h));
      }
    }
  }
}

// ---------------------------------------------------------------------------
// 5) Finalize: decode max accumulator, append pos_s and batch_s
// ---------------------------------------------------------------------------
__global__ void finalize_kernel(const unsigned* __restrict__ outMax,
                                const float* __restrict__ pos_s,
                                float* __restrict__ out) {
  const int i = blockIdx.x * 256 + threadIdx.x;
  const int n0 = BB * MM * COUT;           // 1,048,576 feature outputs
  const int n1 = n0 + BB * MM * 3;         // + pos_s
  const int n2 = n1 + BB * MM;             // + batch_s
  if (i < n0) {
    out[i] = decOrderedF32(outMax[i]);
  } else if (i < n1) {
    out[i] = pos_s[i - n0];
  } else if (i < n2) {
    ((int*)out)[i] = (i - n1) >> 10;       // batch = q / M
  }
}

// ---------------------------------------------------------------------------
extern "C" void kernel_launch(void* const* d_in, const int* in_sizes, int n_in,
                              void* d_out, int out_size, void* d_ws, size_t ws_size,
                              hipStream_t stream) {
  const float* x   = (const float*)d_in[0];
  const float* pos = (const float*)d_in[1];
  // d_in[2] = batch (values derivable analytically; unused)
  const float* W1 = (const float*)d_in[3];
  const float* b1 = (const float*)d_in[4];
  const float* W2 = (const float*)d_in[5];
  const float* b2 = (const float*)d_in[6];
  const float* W3 = (const float*)d_in[7];
  const float* b3 = (const float*)d_in[8];

  char* ws = (char*)d_ws;
  int*      fpsIdx   = (int*)ws;        ws += (size_t)BB * MM * sizeof(int);
  int*      neigh    = (int*)ws;        ws += (size_t)BB * MM * KK * sizeof(int);
  unsigned* maskbits = (unsigned*)ws;   ws += (size_t)BB * MM * sizeof(unsigned);
  float*    pos_s    = (float*)ws;      ws += (size_t)BB * MM * 3 * sizeof(float);
  unsigned* outMax   = (unsigned*)ws;

  fps_kernel<<<BB, 1024, 0, stream>>>(pos, fpsIdx);
  radius_kernel<<<(BB * MM * 32) / 256, 256, 0, stream>>>(pos, fpsIdx, neigh,
                                                          maskbits, pos_s);
  init_kernel<<<(BB * MM * COUT + 255) / 256, 256, 0, stream>>>(outMax);
  mlp_kernel<<<(BB * MM * SLOTS) / 64, 128, 0, stream>>>(
      x, pos, W1, b1, W2, b2, W3, b3, neigh, maskbits, pos_s, outMax);
  const int totalOut = BB * MM * COUT + BB * MM * 3 + BB * MM;
  finalize_kernel<<<(totalOut + 255) / 256, 256, 0, stream>>>(outMax, pos_s,
                                                              (float*)d_out);
}
